// PSI_NoGate_35622458753029
// MI455X (gfx1250) — compile-verified, matrix operand order checked by reading, not run on hardware
//
#include <hip/hip_runtime.h>

// ---------------------------------------------------------------------------
// PSI_NoGate for MI455X (gfx1250): bf16 WMMA GEMMs (TDM-fed, double-buffered)
// + hierarchical scan for the phase/memory recurrences
// ---------------------------------------------------------------------------

typedef __bf16 bf16;
typedef __attribute__((ext_vector_type(8)))  bf16  v8bf;
typedef __attribute__((ext_vector_type(16))) bf16  v16bf;
typedef __attribute__((ext_vector_type(8)))  float v8f;

#define BSZ   4
#define SEQ   8192
#define DIM   512
#define CTXW  2048          // 4*DIM
#define NCH   64            // chunks along S
#define CHLEN 128           // SEQ / NCH
#define ROWS  32768         // BSZ*SEQ

#define BM 128
#define BN 128
#define BK 32

#if __has_builtin(__builtin_amdgcn_tensor_load_to_lds) && \
    __has_builtin(__builtin_amdgcn_s_wait_tensorcnt)
#define PSI_USE_TDM 1
#else
#define PSI_USE_TDM 0
#endif

#if PSI_USE_TDM
typedef unsigned int u32;
typedef __attribute__((ext_vector_type(4))) unsigned int u32x4;
typedef __attribute__((ext_vector_type(4))) int i32x4;
typedef __attribute__((ext_vector_type(8))) int i32x8;

// Issue one TDM descriptor: load a [BN x BK] bf16 tile (row stride = K elems)
// from global into LDS at lds_dst.  D# layout per CDNA5 ISA.
__device__ __forceinline__ void tdm_load_tile_bf16(const bf16* gsrc,
                                                   bf16* lds_dst,
                                                   int K, int Ntot) {
    u32 lds_addr = (u32)(uintptr_t)lds_dst;   // LDS aperture: low 32 bits
    unsigned long long ga = (unsigned long long)(uintptr_t)gsrc;
    u32x4 g0;
    g0[0] = 1u;                                            // count=1, user D#
    g0[1] = lds_addr;                                      // lds_addr
    g0[2] = (u32)ga;                                       // global_addr[31:0]
    g0[3] = (u32)((ga >> 32) & 0x01FFFFFFull) | (2u << 30); // addr[56:32]|type=2
    i32x8 g1;
    g1[0] = 1 << 16;                                       // data_size=2B, mask=0
    g1[1] = (int)(((u32)K & 0xFFFFu) << 16);               // tensor_dim0 lo16
    g1[2] = (int)((((u32)K >> 16) & 0xFFFFu) |             // tensor_dim0 hi16
                  (((u32)Ntot & 0xFFFFu) << 16));          // tensor_dim1 lo16
    g1[3] = (int)((((u32)Ntot >> 16) & 0xFFFFu) |          // tensor_dim1 hi16
                  ((u32)BK << 16));                        // tile_dim0 = 32
    g1[4] = BN;                                            // tile_dim1=128, dim2=0
    g1[5] = K;                                             // dim0_stride lo32
    g1[6] = 0;                                             // stride hi / dim1_stride
    g1[7] = 0;
    i32x4 z4 = {0, 0, 0, 0};
#if __clang_major__ >= 23
    i32x8 z8 = {0, 0, 0, 0, 0, 0, 0, 0};
    __builtin_amdgcn_tensor_load_to_lds(g0, g1, z4, z4, z8, 0);
#else
    __builtin_amdgcn_tensor_load_to_lds(g0, g1, z4, z4, 0);
#endif
}
#endif  // PSI_USE_TDM

__device__ __forceinline__ float gelu_erf(float v) {
    return 0.5f * v * (1.0f + erff(v * 0.70710678118654752f));
}

// ---------------------------------------------------------------------------
// Weight convert: W[K][N] fp32 -> Wt[N][K] bf16 (transposed, K contiguous)
// ---------------------------------------------------------------------------
__global__ void k_convert_transpose(const float* __restrict__ W,
                                    bf16* __restrict__ Wt, int K, int N) {
    int idx = blockIdx.x * blockDim.x + threadIdx.x;
    if (idx >= K * N) return;
    int n = idx / K;
    int k = idx - n * K;
    Wt[idx] = (bf16)W[(size_t)k * N + n];
}

// ---------------------------------------------------------------------------
// WMMA GEMM: C[M,N] = epi(A[M,K] @ Wt^T + bias)  (Wt stored [N][K] bf16)
// EPI: 0 = bias only, 1 = erf-GELU, 2 = sigmoid*5, 3 = bias + residual
// Block: 256 thr = 8 waves (2 M-groups x 4 N-groups); wave tile 64x32.
// Double-buffered LDS; B tiles DMA'd by the Tensor Data Mover (wave 0),
// A tiles converted fp32->bf16 in-flight by all threads.
// ---------------------------------------------------------------------------
template <int EPI>
__global__ __launch_bounds__(256)
void k_gemm_wmma(const float* __restrict__ A, const bf16* __restrict__ Wt,
                 const float* __restrict__ bias, float* __restrict__ C,
                 const float* __restrict__ resid, int M, int N, int K) {
    __shared__ __align__(16) bf16 As[2][BM][BK];
    __shared__ __align__(16) bf16 Bs[2][BN][BK];

    const int tid  = threadIdx.x;
    const int bm   = blockIdx.y, bn = blockIdx.x;
    const int wid  = tid >> 5, lane = tid & 31;
    const int wm   = wid & 1;        // 0..1  -> 64-row group
    const int wn   = wid >> 1;       // 0..3  -> 32-col group
    const int half = lane >> 4;      // K-half / M-half selector
    const int l16  = lane & 15;

    v8f acc[4][2];
#pragma unroll
    for (int mi = 0; mi < 4; ++mi)
#pragma unroll
        for (int ni = 0; ni < 2; ++ni)
#pragma unroll
            for (int e = 0; e < 8; ++e) acc[mi][ni][e] = 0.0f;

    const int ldr = tid >> 1;             // 0..127
    const int ldc = (tid & 1) * 16;       // 0 or 16
    const float* aRow = &A[(size_t)(bm * BM + ldr) * K + ldc];
    const bf16*  bRow = &Wt[(size_t)(bn * BN + ldr) * K + ldc];
    const bf16*  bTile = &Wt[(size_t)(bn * BN) * K];   // tile origin for TDM

    float4 fr[4];
    auto loadA = [&](int kb) {
        const float4* s = reinterpret_cast<const float4*>(aRow + kb);
#pragma unroll
        for (int i = 0; i < 4; ++i) fr[i] = s[i];
    };
    auto storeA = [&](int buf) {
        bf16* dst = &As[buf][ldr][ldc];
#pragma unroll
        for (int i = 0; i < 4; ++i) {
            dst[4 * i + 0] = (bf16)fr[i].x;
            dst[4 * i + 1] = (bf16)fr[i].y;
            dst[4 * i + 2] = (bf16)fr[i].z;
            dst[4 * i + 3] = (bf16)fr[i].w;
        }
    };
#if !PSI_USE_TDM
    auto loadB = [&](int buf, int kb) {
        const uint4* src = reinterpret_cast<const uint4*>(bRow + kb);
        uint4* dst = reinterpret_cast<uint4*>(&Bs[buf][ldr][ldc]);
        dst[0] = src[0];
        dst[1] = src[1];
    };
#else
    (void)bRow;
#endif

    // ---- prologue: stage tile 0 ----
    loadA(0);
    storeA(0);
#if PSI_USE_TDM
    if (wid == 0) {
        tdm_load_tile_bf16(bTile, &Bs[0][0][0], K, N);
        __builtin_amdgcn_s_wait_tensorcnt(0);   // s_wait_tensorcnt 0
    }
#else
    loadB(0, 0);
#endif
    __syncthreads();

    int buf = 0;
    for (int kb = 0; kb < K; kb += BK) {
        const int  nb   = buf ^ 1;
        const bool more = (kb + BK) < K;
        if (more) {
            loadA(kb + BK);                     // global loads overlap WMMA
#if PSI_USE_TDM
            if (wid == 0)                        // async DMA overlaps WMMA
                tdm_load_tile_bf16(bTile + (size_t)(kb + BK),
                                   &Bs[nb][0][0], K, N);
#endif
            __builtin_prefetch(aRow + kb + 2 * BK, 0, 0);  // global_prefetch_b8
        }

        // --- fragments: lanes 0-15 hold K {0..7,16..23}, lanes 16-31 {8..15,24..31}
        v16bf afrag[4], bfrag[2];
#pragma unroll
        for (int mi = 0; mi < 4; ++mi) {
            int row = wm * 64 + mi * 16 + l16;
            v8bf a0 = *reinterpret_cast<const v8bf*>(&As[buf][row][half * 8]);
            v8bf a1 = *reinterpret_cast<const v8bf*>(&As[buf][row][16 + half * 8]);
#pragma unroll
            for (int i = 0; i < 8; ++i) { afrag[mi][i] = a0[i]; afrag[mi][i + 8] = a1[i]; }
        }
#pragma unroll
        for (int ni = 0; ni < 2; ++ni) {
            int col = wn * 32 + ni * 16 + l16;
            v8bf b0 = *reinterpret_cast<const v8bf*>(&Bs[buf][col][half * 8]);
            v8bf b1 = *reinterpret_cast<const v8bf*>(&Bs[buf][col][16 + half * 8]);
#pragma unroll
            for (int i = 0; i < 8; ++i) { bfrag[ni][i] = b0[i]; bfrag[ni][i + 8] = b1[i]; }
        }

#pragma unroll
        for (int mi = 0; mi < 4; ++mi)
#pragma unroll
            for (int ni = 0; ni < 2; ++ni)
                acc[mi][ni] = __builtin_amdgcn_wmma_f32_16x16x32_bf16(
                    false, afrag[mi], false, bfrag[ni], (short)0, acc[mi][ni],
                    false, false);

        if (more) {
            storeA(nb);
#if PSI_USE_TDM
            if (wid == 0) __builtin_amdgcn_s_wait_tensorcnt(0);
#else
            loadB(nb, kb + BK);
#endif
        }
        __syncthreads();
        buf = nb;
    }

    // --- epilogue: D layout = lane -> N (l16), vgpr v -> M = 8*half + v ---
#pragma unroll
    for (int mi = 0; mi < 4; ++mi)
#pragma unroll
        for (int ni = 0; ni < 2; ++ni) {
            int gcol = bn * BN + wn * 32 + ni * 16 + l16;
            float bv = bias[gcol];
#pragma unroll
            for (int v = 0; v < 8; ++v) {
                int grow = bm * BM + wm * 64 + mi * 16 + half * 8 + v;
                float val = acc[mi][ni][v] + bv;
                if (EPI == 1) val = gelu_erf(val);
                else if (EPI == 2) val = 5.0f / (1.0f + __expf(-val));
                else if (EPI == 3) val += resid[(size_t)grow * N + gcol];
                C[(size_t)grow * N + gcol] = val;
            }
        }
}

// ---------------------------------------------------------------------------
// Scan pass 1: per (b, chunk, d) sums of omega*|scale| and magnitude
// grid = B*NCH blocks, 256 threads (2 d's per thread)
// ---------------------------------------------------------------------------
__global__ void k_scan_part1(const float* __restrict__ omega,
                             const float* __restrict__ mag,
                             const float* __restrict__ scale,
                             float* __restrict__ pOm, float* __restrict__ pMg) {
    int b = blockIdx.x >> 6, ch = blockIdx.x & 63;
#pragma unroll
    for (int j = 0; j < 2; ++j) {
        int d = threadIdx.x + j * 256;
        float sc = fabsf(scale[d]);
        size_t base = ((size_t)b * SEQ + ch * CHLEN) * DIM + d;
        float so = 0.f, sm = 0.f;
        for (int s = 0; s < CHLEN; ++s) {
            size_t i = base + (size_t)s * DIM;
            so += omega[i] * sc;
            sm += mag[i];
        }
        size_t p = ((size_t)b * NCH + ch) * DIM + d;
        pOm[p] = so;
        pMg[p] = sm;
    }
}

// Exclusive scan over chunks for two arrays [B][NCH][D], in place.
__global__ void k_scan_chunks(float* __restrict__ p0, float* __restrict__ p1) {
    int idx = blockIdx.x * blockDim.x + threadIdx.x;   // over B*D = 2048
    if (idx >= BSZ * DIM) return;
    int b = idx >> 9, d = idx & (DIM - 1);
    size_t base = (size_t)b * NCH * DIM + d;
    float a0 = 0.f, a1 = 0.f;
    for (int ch = 0; ch < NCH; ++ch) {
        size_t o = base + (size_t)ch * DIM;
        float t0 = p0[o], t1 = p1[o];
        p0[o] = a0; p1[o] = a1;
        a0 += t0;  a1 += t1;
    }
}

// ---------------------------------------------------------------------------
// Scan pass 2: with phi chunk-bases known, chunk sums of wc*cos/wc*sin
// ---------------------------------------------------------------------------
__global__ void k_scan_part2(const float* __restrict__ omega,
                             const float* __restrict__ mag,
                             const float* __restrict__ x,
                             const float* __restrict__ phi_init,
                             const float* __restrict__ scale,
                             const float* __restrict__ pOm,
                             float* __restrict__ pR, float* __restrict__ pI) {
    int b = blockIdx.x >> 6, ch = blockIdx.x & 63;
#pragma unroll
    for (int j = 0; j < 2; ++j) {
        int d = threadIdx.x + j * 256;
        float sc = fabsf(scale[d]);
        size_t base = ((size_t)b * SEQ + ch * CHLEN) * DIM + d;
        size_t p = ((size_t)b * NCH + ch) * DIM + d;
        float pb = pOm[p];
        float lo = 0.f, sr = 0.f, si = 0.f;
        for (int s = 0; s < CHLEN; ++s) {
            size_t i = base + (size_t)s * DIM;
            lo += omega[i] * sc;
            float phi = phi_init[i] + pb + lo;
            float sn, cs;
            sincosf(phi, &sn, &cs);
            float wc = mag[i] * x[i];
            sr += wc * cs;
            si += wc * sn;
        }
        pR[p] = sr;
        pI[p] = si;
    }
}

// ---------------------------------------------------------------------------
// Scan final: build ctx = [x*cos, x*sin, ret_r, ret_i]  [B,S,4D]
// ---------------------------------------------------------------------------
__global__ void k_scan_final(const float* __restrict__ omega,
                             const float* __restrict__ mag,
                             const float* __restrict__ x,
                             const float* __restrict__ phi_init,
                             const float* __restrict__ qadd,
                             const float* __restrict__ scale,
                             const float* __restrict__ pOm,
                             const float* __restrict__ pMg,
                             const float* __restrict__ pR,
                             const float* __restrict__ pI,
                             float* __restrict__ ctx) {
    int b = blockIdx.x >> 6, ch = blockIdx.x & 63;
#pragma unroll
    for (int j = 0; j < 2; ++j) {
        int d = threadIdx.x + j * 256;
        float sc = fabsf(scale[d]);
        size_t base = ((size_t)b * SEQ + ch * CHLEN) * DIM + d;
        size_t p = ((size_t)b * NCH + ch) * DIM + d;
        float pb = pOm[p], mb = pMg[p], rb = pR[p], ib = pI[p];
        float lo = 0.f, lm = 0.f, lr = 0.f, li = 0.f;
        for (int s = 0; s < CHLEN; ++s) {
            size_t i = base + (size_t)s * DIM;
            lo += omega[i] * sc;
            float phi = phi_init[i] + pb + lo;
            float mg = mag[i];
            lm += mg;
            float xv = x[i];
            float sn, cs;
            sincosf(phi, &sn, &cs);
            float wc = mg * xv;
            lr += wc * cs;
            li += wc * sn;
            float rs  = rsqrtf(mb + lm + 1e-8f);
            float mr  = (rb + lr) * rs;
            float mi2 = (ib + li) * rs;
            float pq = phi + qadd[i];
            float sq, cq;
            sincosf(pq, &sq, &cq);
            float rr = mr * cq + mi2 * sq;
            float ri = mi2 * cq - mr * sq;
            size_t c0 = ((size_t)b * SEQ + ch * CHLEN + s) * (size_t)CTXW + d;
            ctx[c0]            = xv * cs;
            ctx[c0 + DIM]      = xv * sn;
            ctx[c0 + 2 * DIM]  = rr;
            ctx[c0 + 3 * DIM]  = ri;
        }
    }
}

// ---------------------------------------------------------------------------
// LayerNorm over 4D = 2048; one block per row
// ---------------------------------------------------------------------------
__global__ __launch_bounds__(256)
void k_layernorm(const float* __restrict__ ctx, const float* __restrict__ g,
                 const float* __restrict__ bta, float* __restrict__ out) {
    int row = blockIdx.x;
    const float* p = ctx + (size_t)row * CTXW;
    float s1 = 0.f, s2 = 0.f;
    for (int j = threadIdx.x; j < CTXW; j += 256) {
        float v = p[j];
        s1 += v;
        s2 += v * v;
    }
    __shared__ float r1[256], r2[256];
    r1[threadIdx.x] = s1;
    r2[threadIdx.x] = s2;
    __syncthreads();
    for (int off = 128; off > 0; off >>= 1) {
        if (threadIdx.x < off) {
            r1[threadIdx.x] += r1[threadIdx.x + off];
            r2[threadIdx.x] += r2[threadIdx.x + off];
        }
        __syncthreads();
    }
    float mu   = r1[0] * (1.0f / CTXW);
    float var  = r2[0] * (1.0f / CTXW) - mu * mu;
    float rstd = rsqrtf(var + 1e-5f);
    float* o = out + (size_t)row * CTXW;
    for (int j = threadIdx.x; j < CTXW; j += 256)
        o[j] = (p[j] - mu) * rstd * g[j] + bta[j];
}

// ---------------------------------------------------------------------------
// Launch
// ---------------------------------------------------------------------------
static void launch_gemm(int epi, const float* A, const bf16* Wt, const float* bias,
                        float* C, const float* resid, int M, int N, int K,
                        hipStream_t s) {
    dim3 g(N / BN, M / BM), b(256);
    switch (epi) {
        case 0: k_gemm_wmma<0><<<g, b, 0, s>>>(A, Wt, bias, C, resid, M, N, K); break;
        case 1: k_gemm_wmma<1><<<g, b, 0, s>>>(A, Wt, bias, C, resid, M, N, K); break;
        case 2: k_gemm_wmma<2><<<g, b, 0, s>>>(A, Wt, bias, C, resid, M, N, K); break;
        default: k_gemm_wmma<3><<<g, b, 0, s>>>(A, Wt, bias, C, resid, M, N, K); break;
    }
}

extern "C" void kernel_launch(void* const* d_in, const int* in_sizes, int n_in,
                              void* d_out, int out_size, void* d_ws, size_t ws_size,
                              hipStream_t stream) {
    (void)in_sizes; (void)n_in; (void)out_size; (void)ws_size;

    const float* x       = (const float*)d_in[0];
    const float* W_omega = (const float*)d_in[1];
    const float* b_omega = (const float*)d_in[2];
    const float* W_p1    = (const float*)d_in[3];
    const float* b_p1    = (const float*)d_in[4];
    const float* W_p2    = (const float*)d_in[5];
    const float* b_p2    = (const float*)d_in[6];
    const float* scale   = (const float*)d_in[7];
    const float* W_mag   = (const float*)d_in[8];
    const float* b_mag   = (const float*)d_in[9];
    const float* W_q     = (const float*)d_in[10];
    const float* b_q     = (const float*)d_in[11];
    const float* ln_g    = (const float*)d_in[12];
    const float* ln_b    = (const float*)d_in[13];
    const float* W_o1    = (const float*)d_in[14];
    const float* b_o1    = (const float*)d_in[15];
    const float* W_o2    = (const float*)d_in[16];
    const float* b_o2    = (const float*)d_in[17];

    constexpr size_t SZ   = (size_t)ROWS * DIM;       // 16.78M floats
    constexpr size_t PART = (size_t)BSZ * NCH * DIM;  // 131072 floats

    float* fw    = (float*)d_ws;
    float* omega = fw;                 // [ROWS,DIM]
    float* magn  = fw + 1 * SZ;
    float* g1    = fw + 2 * SZ;
    float* phi0  = fw + 3 * SZ;
    float* qadd  = fw + 4 * SZ;
    float* ctx   = fw + 5 * SZ;        // [ROWS,4D] (4*SZ)
    float* pOm   = fw + 9 * SZ;
    float* pMg   = pOm + PART;
    float* pR    = pMg + PART;
    float* pI    = pR + PART;
    float* ln    = omega;              // alias: 4*SZ fits in region A (5*SZ)
    float* h     = ctx;                // alias: 2*SZ fits in ctx region
    bf16* wOm = (bf16*)(pI + PART);
    bf16* wP1 = wOm + (size_t)DIM * DIM;
    bf16* wP2 = wP1 + (size_t)DIM * DIM;
    bf16* wMg = wP2 + (size_t)DIM * DIM;
    bf16* wQ  = wMg + (size_t)DIM * DIM;
    bf16* wO1 = wQ  + (size_t)DIM * DIM;              // [1024][2048]
    bf16* wO2 = wO1 + (size_t)CTXW * (2 * DIM);       // [512][1024]

    auto conv = [&](const float* W, bf16* Wt, int K, int N) {
        int total = K * N;
        k_convert_transpose<<<(total + 255) / 256, 256, 0, stream>>>(W, Wt, K, N);
    };
    conv(W_omega, wOm, DIM, DIM);
    conv(W_p1,   wP1, DIM, DIM);
    conv(W_p2,   wP2, DIM, DIM);
    conv(W_mag,  wMg, DIM, DIM);
    conv(W_q,    wQ,  DIM, DIM);
    conv(W_o1,   wO1, CTXW, 2 * DIM);
    conv(W_o2,   wO2, 2 * DIM, DIM);

    // Stage 1: input projections (WMMA bf16, fp32 accum)
    launch_gemm(0, x,  wOm, b_omega, omega, nullptr, ROWS, DIM, DIM, stream);
    launch_gemm(2, x,  wMg, b_mag,   magn,  nullptr, ROWS, DIM, DIM, stream);
    launch_gemm(1, x,  wP1, b_p1,    g1,    nullptr, ROWS, DIM, DIM, stream);
    launch_gemm(0, g1, wP2, b_p2,    phi0,  nullptr, ROWS, DIM, DIM, stream);
    launch_gemm(0, x,  wQ,  b_q,     qadd,  nullptr, ROWS, DIM, DIM, stream);

    // Stage 2: hierarchical scans along S
    dim3 sg(BSZ * NCH), sb(256);
    k_scan_part1<<<sg, sb, 0, stream>>>(omega, magn, scale, pOm, pMg);
    k_scan_chunks<<<(BSZ * DIM + 255) / 256, 256, 0, stream>>>(pOm, pMg);
    k_scan_part2<<<sg, sb, 0, stream>>>(omega, magn, x, phi0, scale, pOm, pR, pI);
    k_scan_chunks<<<(BSZ * DIM + 255) / 256, 256, 0, stream>>>(pR, pI);
    k_scan_final<<<sg, sb, 0, stream>>>(omega, magn, x, phi0, qadd, scale,
                                        pOm, pMg, pR, pI, ctx);

    // Stage 3: LayerNorm + output MLP
    k_layernorm<<<ROWS, 256, 0, stream>>>(ctx, ln_g, ln_b, ln);
    launch_gemm(1, ln, wO1, b_o1, h, nullptr, ROWS, 2 * DIM, CTXW, stream);
    launch_gemm(3, h,  wO2, b_o2, (float*)d_out, x, ROWS, DIM, 2 * DIM, stream);
}
